// LogicalReindexBranch_77481210020132
// MI455X (gfx1250) — matrix operand order repaired and checked by reading, not compiled, not run on hardware
//
#include <hip/hip_runtime.h>
#include <hip/hip_bf16.h>

// CDNA5 (gfx1250) fused kernel: TDM-staged A -> feature build -> Linear(4,128)
// via f32 WMMA -> GEMM(128x128) via f16 WMMA -> fused LN -> scores via f16
// WMMA -> multiplicative Sinkhorn (== log-domain LSE Sinkhorn, then exp).

typedef __attribute__((ext_vector_type(16))) _Float16 v16h;
typedef __attribute__((ext_vector_type(8)))  _Float16 v8h;
typedef __attribute__((ext_vector_type(8)))  float    v8f;
typedef __attribute__((ext_vector_type(2)))  float    v2f;
typedef __attribute__((ext_vector_type(4)))  unsigned int u32x4;
typedef __attribute__((ext_vector_type(8)))  int      i32x8;
typedef __attribute__((ext_vector_type(4)))  int      i32x4;

#define GB     4          // batches per workgroup
#define NN     27
#define HIDD   128
#define STR    136        // f16 LDS row stride (128 + 8) -> 16B aligned, conflict-poor
#define SROW   33         // f32 Sinkhorn matrix row stride (32 + 1)
#define ITERS  20
#define LNEPS  1e-5f

#if __has_builtin(__builtin_amdgcn_tensor_load_to_lds) && \
    __has_builtin(__builtin_amdgcn_s_wait_tensorcnt)
#define USE_TDM 1
#else
#define USE_TDM 0
#endif

union V16U { v16h v; v8h h[2]; };

// A-fragment (16x32 f16) per ISA 7.12.2: lanes 0-15 rows M=0..15 hold K=kb+0..7
// and kb+16..23; lanes 16-31 same rows hold K=kb+8..15 and kb+24..31.
__device__ __forceinline__ v16h ld_afrag(const _Float16* base, int rowbase, int kbase, int lane) {
  const int hh = lane >> 4;
  const int r  = rowbase + (lane & 15);
  const _Float16* p = base + r * STR + kbase + (hh ? 8 : 0);
  V16U u;
  u.h[0] = *(const v8h*)(p);
  u.h[1] = *(const v8h*)(p + 16);
  return u.v;
}

// B-fragment (32x16 f16): lane holds column n = nbase + lane%16 with
// K = kbase + (lane<16 ? 0..7,16..23 : 8..15,24..31) (matches A); base is [n][k] layout.
__device__ __forceinline__ v16h ld_bfrag(const _Float16* base, int nbase, int kbase, int lane) {
  const int hh = lane >> 4;
  const int n  = nbase + (lane & 15);
  const _Float16* p = base + n * STR + kbase + (hh ? 8 : 0);
  V16U u;
  u.h[0] = *(const v8h*)(p);
  u.h[1] = *(const v8h*)(p + 16);
  return u.v;
}

__global__ void __launch_bounds__(256)
sinkhorn_fused(const float* __restrict__ A, const float* __restrict__ m,
               const float* __restrict__ tau_r,
               const float* __restrict__ ln1_g, const float* __restrict__ ln1_b,
               const float* __restrict__ W1, const float* __restrict__ b1,
               const float* __restrict__ W2, const float* __restrict__ b2,
               const float* __restrict__ ln2_g, const float* __restrict__ ln2_b,
               const float* __restrict__ protos, float* __restrict__ out)
{
  extern __shared__ char smem[];
  float* A_s = (float*)smem;                         // GB*729 (offset 0 in LDS alloc)
  float* S   = A_s + GB * 729;                       // GB*32*SROW
  float* W1T = S + GB * 32 * SROW;                   // 512  (W1 transposed: [n][k], k=0..3)
  float* b1s = W1T + 512;                            // 128
  float* b2s = b1s + 128;                            // 128
  float* g2s = b2s + 128;                            // 128
  float* t2s = g2s + 128;                            // 128
  float* Xs  = t2s + 128;                            // 128*4 (LN'd features, pad rows zero)
  _Float16* H1  = (_Float16*)(Xs + 512);             // 128*STR (f16)
  _Float16* Hn  = H1  + 128 * STR;                   // 128*STR
  _Float16* W2T = Hn  + 128 * STR;                   // 128*STR, [n][k]
  _Float16* PT  = W2T + 128 * STR;                   // 32*STR,  [n][k]

  const int tid  = threadIdx.x;
  const int lane = tid & 31;
  const int wave = tid >> 5;
  const int wg   = blockIdx.x;

  // ---------------- phase 0: stage A via Tensor Data Mover (DMA to LDS) ----------------
#if USE_TDM
  if (wave == 0) {
    // D# per ISA 8.3/8.4: 1-row tile of GB*729 f32 elements, global -> LDS offset 0.
    unsigned long long ga = (unsigned long long)(const void*)(A + wg * (GB * 729));
    unsigned lds_off = (unsigned)(unsigned long long)(const void*)A_s; // LDS byte address from low 32 bits of pointer
    u32x4 g0;
    g0[0] = 1u;                                            // count=1, user descriptor
    g0[1] = lds_off;                                       // lds_addr (bytes)
    g0[2] = (unsigned)(ga & 0xffffffffu);                  // global_addr[31:0]
    g0[3] = (unsigned)((ga >> 32) & 0x01ffffffu)           // global_addr[56:32]
          | 0x80000000u;                                   // type=2 ("image") in bits 127:126
    i32x8 g1;
    g1[0] = (int)(2u << 16);                               // data_size=2 -> 4 bytes
    g1[1] = (int)((unsigned)(GB * 729) << 16);             // tensor_dim0[15:0] @bits63:48
    g1[2] = (int)(1u << 16);                               // tensor_dim0[31:16]=0, tensor_dim1[15:0]=1
    g1[3] = (int)((unsigned)(GB * 729) << 16);             // tensor_dim1[31:16]=0, tile_dim0=2916
    g1[4] = 1;                                             // tile_dim1=1, tile_dim2=0
    g1[5] = (int)(GB * 729);                               // tensor_dim0_stride[31:0]
    g1[6] = 0;                                             // stride hi / dim1_stride lo
    g1[7] = 0;
    i32x4 z4 = {0, 0, 0, 0};
#if __clang_major__ >= 23
    i32x8 z8 = {0, 0, 0, 0, 0, 0, 0, 0};
    __builtin_amdgcn_tensor_load_to_lds(g0, g1, z4, z4, z8, 0);
#else
    __builtin_amdgcn_tensor_load_to_lds(g0, g1, z4, z4, 0);
#endif
    __builtin_amdgcn_s_wait_tensorcnt(0);
  }
#else
  for (int i = tid; i < GB * 729; i += 256) A_s[i] = A[wg * (GB * 729) + i];
#endif

  // ---------------- phase 0b: cooperative staging of weights (overlaps TDM) ----------------
  for (int i = tid; i < 128 * 128; i += 256) {          // W2 row-major [k][n] -> W2T [n][k] f16
    int k = i >> 7, n = i & 127;
    W2T[n * STR + k] = (_Float16)W2[i];
  }
  for (int i = tid; i < 32 * 128; i += 256) {           // protos [n][k] f16, rows 27..31 zero
    int n = i >> 7, k = i & 127;
    PT[n * STR + k] = (n < NN) ? (_Float16)protos[n * HIDD + k] : (_Float16)0.0f;
  }
  for (int i = tid; i < 512; i += 256) {                // W1 (4,128) -> W1T [n][k]
    int n = i >> 2, k = i & 3;
    W1T[i] = W1[k * HIDD + n];
  }
  if (tid < 128) {
    b1s[tid] = b1[tid]; b2s[tid] = b2[tid];
    g2s[tid] = ln2_g[tid]; t2s[tid] = ln2_b[tid];
  }
  for (int i = tid; i < GB * 5 * 4; i += 256) {         // zero X pad rows 27..31 per batch
    int b = i / 20, rem = i % 20, r = rem >> 2, k = rem & 3;
    Xs[(b * 32 + 27 + r) * 4 + k] = 0.f;
  }
  __syncthreads();

  // ---------------- phase 1: feature build + LN(4) -> Xs ----------------
  if (tid < GB * NN) {
    int b = tid / NN, n = tid - b * NN;
    const float* Ar = A_s + b * 729 + n * 27;
    float rs = 0.f, m1 = -1e30f, m2 = -1e30f;
    #pragma unroll
    for (int c = 0; c < 27; ++c) {
      float v = Ar[c];
      rs += v;
      if (c != n) {
        if (v > m1) { m2 = m1; m1 = v; }
        else if (v > m2) { m2 = v; }
      }
    }
    float f0 = log1pf(rs), f1 = m1, f2 = m2, f3 = m[(wg * GB + b) * NN + n];
    float mu  = 0.25f * (f0 + f1 + f2 + f3);
    float d0 = f0 - mu, d1 = f1 - mu, d2 = f2 - mu, d3 = f3 - mu;
    float var = 0.25f * (d0 * d0 + d1 * d1 + d2 * d2 + d3 * d3);
    float rsd = rsqrtf(var + LNEPS);
    float* xr = Xs + (b * 32 + n) * 4;
    xr[0] = d0 * rsd * ln1_g[0] + ln1_b[0];
    xr[1] = d1 * rsd * ln1_g[1] + ln1_b[1];
    xr[2] = d2 * rsd * ln1_g[2] + ln1_b[2];
    xr[3] = d3 * rsd * ln1_g[3] + ln1_b[3];
  }
  __syncthreads();

  const int rowbase = wave * 16;        // each wave owns one 16-row M-tile (8 waves x 16 = 128)
  const int hh = lane >> 4;
  const int nl = lane & 15;

  v8f zero = {0.f, 0.f, 0.f, 0.f, 0.f, 0.f, 0.f, 0.f};

  // ---------------- phase 1b: H1 = relu(X @ W1 + b1) via v_wmma_f32_16x16x4_f32 ----------------
  {
    // A-frag (16x4 f32): VGPR0 = K=(hh?2:0), VGPR1 = K+1, rows = lanes%16
    v2f a2 = *(const v2f*)(Xs + ((rowbase + nl) << 2) + (hh ? 2 : 0));
    #pragma unroll
    for (int t = 0; t < 8; ++t) {
      int colb = t * 16 + nl;
      // B-frag (4x16 f32): VGPR0 = K=(hh?2:0) col=lane%16, VGPR1 = K+1
      v2f bv = *(const v2f*)(W1T + (colb << 2) + (hh ? 2 : 0));
      v8f h = __builtin_amdgcn_wmma_f32_16x16x4_f32(
          false, a2, false, bv, (short)0, zero, false, false);
      float bb = b1s[colb];
      #pragma unroll
      for (int i = 0; i < 8; ++i) {
        int row = rowbase + i + 8 * hh;
        H1[row * STR + colb] = (_Float16)fmaxf(h[i] + bb, 0.f);
      }
    }
  }
  __syncthreads();

  // ---------------- phase 2: H1 @ W2 + b2 via f16 WMMA, then LN(128) in registers ----------------
  v8f acc[8];
  #pragma unroll
  for (int t = 0; t < 8; ++t) acc[t] = zero;

  #pragma unroll
  for (int kk = 0; kk < 4; ++kk) {
    v16h af = ld_afrag(H1, rowbase, kk * 32, lane);
    #pragma unroll
    for (int t = 0; t < 8; ++t) {
      v16h bf = ld_bfrag(W2T, t * 16, kk * 32, lane);
      acc[t] = __builtin_amdgcn_wmma_f32_16x16x32_f16(
          false, af, false, bf, (short)0, acc[t], false, false);
    }
  }
  // bias
  #pragma unroll
  for (int t = 0; t < 8; ++t) {
    float bb = b2s[t * 16 + nl];
    #pragma unroll
    for (int i = 0; i < 8; ++i) acc[t][i] += bb;
  }
  // per-row mean/var: 8 accum partials + 4-step shuffle reduce within 16-lane halves
  float mus[8], rss[8];
  #pragma unroll
  for (int i = 0; i < 8; ++i) {
    float ss = 0.f, qq = 0.f;
    #pragma unroll
    for (int t = 0; t < 8; ++t) { float v = acc[t][i]; ss += v; qq += v * v; }
    #pragma unroll
    for (int msk = 1; msk <= 8; msk <<= 1) {
      ss += __shfl_xor(ss, msk, 32);
      qq += __shfl_xor(qq, msk, 32);
    }
    float mu = ss * (1.f / 128.f);
    float vv = qq * (1.f / 128.f) - mu * mu;
    mus[i] = mu;
    rss[i] = rsqrtf(fmaxf(vv, 0.f) + LNEPS);
  }
  // normalize, scale/shift, store f16 Hn
  #pragma unroll
  for (int i = 0; i < 8; ++i) {
    int row = rowbase + i + 8 * hh;
    #pragma unroll
    for (int t = 0; t < 8; ++t) {
      int col = t * 16 + nl;
      float v = (acc[t][i] - mus[i]) * rss[i] * g2s[col] + t2s[col];
      Hn[row * STR + col] = (_Float16)v;
    }
  }
  __syncthreads();

  // ---------------- phase 3: scores = Hn @ protos^T via f16 WMMA; write exp(score/tau) transposed ----------------
  const float invtau = 1.0f / tau_r[0];
  v8f sc[2];
  sc[0] = zero; sc[1] = zero;
  #pragma unroll
  for (int kk = 0; kk < 4; ++kk) {
    v16h af = ld_afrag(Hn, rowbase, kk * 32, lane);
    #pragma unroll
    for (int t = 0; t < 2; ++t) {
      v16h bf = ld_bfrag(PT, t * 16, kk * 32, lane);
      sc[t] = __builtin_amdgcn_wmma_f32_16x16x32_f16(
          false, af, false, bf, (short)0, sc[t], false, false);
    }
  }
  #pragma unroll
  for (int t = 0; t < 2; ++t) {
    int kproto = t * 16 + nl;                 // slot-prototype index (transposed axis)
    #pragma unroll
    for (int i = 0; i < 8; ++i) {
      int row = rowbase + i + 8 * hh;         // global M row in [0,128)
      int b = row >> 5, nloc = row & 31;
      // log_alpha[b][kproto][nloc] = score/tau ; start multiplicative Sinkhorn from exp()
      S[b * (32 * SROW) + kproto * SROW + nloc] = expf(sc[t][i] * invtau);
    }
  }
  __syncthreads();

  // ---------------- phase 4: multiplicative Sinkhorn (== log-domain w/ LSE, then exp) ----------------
  {
    int sb = tid >> 6;            // batch within workgroup (0..3)
    int j  = tid & 63;            // row/col owner
    float* Sb = S + sb * (32 * SROW);
    bool act = (j < NN);
    for (int it = 0; it < ITERS; ++it) {
      if (act) {                  // normalize row j (axis -1)
        float* r = Sb + j * SROW;
        float t = 0.f;
        #pragma unroll
        for (int c = 0; c < NN; ++c) t += r[c];
        float inv = 1.f / t;
        #pragma unroll
        for (int c = 0; c < NN; ++c) r[c] *= inv;
      }
      __syncthreads();
      if (act) {                  // normalize column j (axis -2)
        float t = 0.f;
        #pragma unroll
        for (int i2 = 0; i2 < NN; ++i2) t += Sb[i2 * SROW + j];
        float inv = 1.f / t;
        #pragma unroll
        for (int i2 = 0; i2 < NN; ++i2) Sb[i2 * SROW + j] *= inv;
      }
      __syncthreads();
    }
  }

  // ---------------- phase 5: coalesced output ----------------
  for (int i = tid; i < GB * 729; i += 256) {
    int b = i / 729, rem = i - b * 729;
    int r = rem / 27, c = rem - r * 27;
    out[wg * (GB * 729) + i] = S[b * (32 * SROW) + r * SROW + c];
  }
}

extern "C" void kernel_launch(void* const* d_in, const int* in_sizes, int n_in,
                              void* d_out, int out_size, void* d_ws, size_t ws_size,
                              hipStream_t stream) {
  const float* A      = (const float*)d_in[0];
  const float* m      = (const float*)d_in[1];
  const float* tau_r  = (const float*)d_in[2];
  const float* ln1_g  = (const float*)d_in[3];
  const float* ln1_b  = (const float*)d_in[4];
  const float* W1     = (const float*)d_in[5];
  const float* b1     = (const float*)d_in[6];
  const float* W2     = (const float*)d_in[7];
  const float* b2     = (const float*)d_in[8];
  const float* ln2_g  = (const float*)d_in[9];
  const float* ln2_b  = (const float*)d_in[10];
  const float* protos = (const float*)d_in[11];
  float* out = (float*)d_out;

  const int Bn   = in_sizes[0] / (NN * NN);   // 32768
  const int grid = Bn / GB;                   // 8192 workgroups

  const size_t smem =
      (size_t)(GB * 729 + GB * 32 * SROW + 512 + 4 * 128 + 512) * sizeof(float) +
      (size_t)(3 * 128 * STR + 32 * STR) * sizeof(_Float16);

  hipFuncSetAttribute((const void*)sinkhorn_fused,
                      hipFuncAttributeMaxDynamicSharedMemorySize, (int)smem);
  sinkhorn_fused<<<grid, 256, smem, stream>>>(A, m, tau_r, ln1_g, ln1_b,
                                              W1, b1, W2, b2, ln2_g, ln2_b,
                                              protos, out);
}
